// FeatureMatching_74045236183111
// MI455X (gfx1250) — matrix operand order repaired and checked by reading, not compiled, not run on hardware
//
#include <hip/hip_runtime.h>
#include <math.h>

typedef __attribute__((ext_vector_type(16))) _Float16 v16h;
typedef __attribute__((ext_vector_type(8)))  float    v8f;

#define HW 4096

#ifndef __has_builtin
#define __has_builtin(x) 0
#endif
#if __has_builtin(__builtin_amdgcn_tensor_load_to_lds)
#define HAVE_TDM 1
#else
#define HAVE_TDM 0
#endif

// ---------------------------------------------------------------------------
// WMMA core: D = A(16x32,f16) * B(32x16,f16) + C(16x16,f32), wave32.
// Fragment packing used here (contiguous k, not the ISA table order):
//   A: lane m=l&15, hi=l>>4; K = 16*hi + j
//   B: lane n=l&15, K = 16*hi + j
//   C/D: vgpr i -> M = i + 8*hi, N = l&15
// ---------------------------------------------------------------------------
__device__ inline v8f wmma16(v16h a, v16h b, v8f c) {
  return __builtin_amdgcn_wmma_f32_16x16x32_f16(false, a, false, b, (short)0, c,
                                                false, false);
}

// A fragment from f16 LDS tile laid out [m][k] (m*32+k)
__device__ inline v16h frag_a_h(const _Float16* As) {
  const int l = threadIdx.x & 31, hi = l >> 4, lo = l & 15;
  v16h a;
#pragma unroll
  for (int j = 0; j < 16; ++j) {
    const int ka = 16 * hi + j;
    a[j] = As[lo * 32 + ka];
  }
  return a;
}
// B fragment from f16 LDS tile laid out [k][64], column block ncol0
__device__ inline v16h frag_b_h64(const _Float16* Bs, int ncol0) {
  const int l = threadIdx.x & 31, hi = l >> 4, lo = l & 15;
  v16h b;
#pragma unroll
  for (int j = 0; j < 16; ++j) b[j] = Bs[(16 * hi + j) * 64 + ncol0 + lo];
  return b;
}
// A fragment from f32 LDS tile laid out [k][16] (TDM destination layout)
__device__ inline v16h frag_a_f32t(const float* Af) {
  const int l = threadIdx.x & 31, hi = l >> 4, lo = l & 15;
  v16h a;
#pragma unroll
  for (int j = 0; j < 16; ++j) {
    const int ka = 16 * hi + j;
    a[j] = (_Float16)Af[ka * 16 + lo];
  }
  return a;
}
// B fragment from f32 LDS tile laid out [n][32], row block nrow0
__device__ inline v16h frag_b_f32n(const float* Bf, int nrow0) {
  const int l = threadIdx.x & 31, hi = l >> 4, lo = l & 15;
  v16h b;
#pragma unroll
  for (int j = 0; j < 16; ++j) b[j] = (_Float16)Bf[(nrow0 + lo) * 32 + 16 * hi + j];
  return b;
}

__device__ inline float silu_f(float v) { return v / (1.f + __expf(-v)); }

// ---------------------------------------------------------------------------
// Tensor Data Mover: async 2D tile (tile_w x tile_h elements of 4B) from
// global (row stride = stride_elems) into LDS at lds_off.  D# layout per
// CDNA5 ISA 8.3/8.4 (group0: count/lds/global/type, group1: data_size,
// tensor dims, tile dims, dim0 stride).  Tracked by TENSORcnt.
// ---------------------------------------------------------------------------
#if HAVE_TDM
typedef unsigned int tdm_v4u __attribute__((ext_vector_type(4)));
typedef int tdm_v8i __attribute__((ext_vector_type(8)));
typedef int tdm_v4i __attribute__((ext_vector_type(4)));

__device__ inline void tdm_load_2d_f32(unsigned lds_off, const void* gptr,
                                       unsigned tile_w, unsigned tile_h,
                                       unsigned stride_elems) {
  const unsigned long long ga = (unsigned long long)(uintptr_t)gptr;
  const unsigned tw = 1u << 20, th = 1u << 20;  // tensor dims: tiles in-bounds
  tdm_v4u d0;
  d0[0] = 1u;                                   // count=1, user descriptor
  d0[1] = lds_off;                              // lds_addr (bytes)
  d0[2] = (unsigned)ga;                         // global_addr[31:0]
  d0[3] = (unsigned)((ga >> 32) & 0x01ffffffull) | 0x80000000u;  // type=2
  tdm_v8i d1;
  d1[0] = (int)(2u << 16);                      // data_size = 4 bytes
  d1[1] = (int)((tw & 0xffffu) << 16);          // tensor_dim0[15:0]
  d1[2] = (int)(((tw >> 16) & 0xffffu) | ((th & 0xffffu) << 16));
  d1[3] = (int)(((th >> 16) & 0xffffu) | ((tile_w & 0xffffu) << 16));
  d1[4] = (int)(tile_h & 0xffffu);              // tile_dim1 (tile_dim2 = 0)
  d1[5] = (int)stride_elems;                    // tensor_dim0_stride[31:0]
  d1[6] = 0;
  d1[7] = 0;
  tdm_v4i d2 = {0, 0, 0, 0};
  tdm_v4i d3 = {0, 0, 0, 0};
#if defined(__clang_major__) && __clang_major__ >= 23
  tdm_v8i d4 = {0, 0, 0, 0, 0, 0, 0, 0};
  __builtin_amdgcn_tensor_load_to_lds(d0, d1, d2, d3, d4, 0);
#else
  __builtin_amdgcn_tensor_load_to_lds(d0, d1, d2, d3, 0);
#endif
}
#endif  // HAVE_TDM

// ---------------------------------------------------------------------------
// BN folding: s = g * rsqrt(v + eps), t = b - m * s
// ---------------------------------------------------------------------------
__global__ void fm_prep_bn(const float* g, const float* b, const float* m,
                           const float* v, float* s, float* t, int n) {
  int i = blockIdx.x * blockDim.x + threadIdx.x;
  if (i < n) {
    float sv = g[i] * rsqrtf(v[i] + 1e-5f);
    s[i] = sv;
    t[i] = b[i] - m[i] * sv;
  }
}

// ---------------------------------------------------------------------------
// 1x1 conv as GEMM with TDM-staged tiles. One wave per 16(pixel) x 64(outC)
// block; A fragment reused across 4 WMMAs. imgs 0,1 from x0; 2,3 from x1.
// LDS: Af = 32(k) x 16(m) f32 (TDM tile), Bf = 64(n) x 32(k) f32 (TDM tile).
// ---------------------------------------------------------------------------
__global__ void fm_conv1x1(const float* __restrict__ x0,
                           const float* __restrict__ x1,
                           const float* __restrict__ w,
                           const float* __restrict__ s,
                           const float* __restrict__ t, float* __restrict__ y,
                           int inC, int outC) {
  __shared__ float Af[32 * 16];
  __shared__ float Bf[64 * 32];
  const int l   = threadIdx.x;
  const int m0  = blockIdx.x * 16;
  const int n0  = blockIdx.y * 64;
  const int img = blockIdx.z;
  const float* xi = (img < 2) ? (x0 + (size_t)img * inC * HW)
                              : (x1 + (size_t)(img - 2) * inC * HW);
  float* yi = y + (size_t)img * outC * HW;

  v8f acc[4];
#pragma unroll
  for (int nt = 0; nt < 4; ++nt)
#pragma unroll
    for (int i = 0; i < 8; ++i) acc[nt][i] = 0.f;

  for (int k0 = 0; k0 < inC; k0 += 32) {
    if (k0 + 32 < inC)
      __builtin_prefetch(xi + (size_t)(k0 + 32) * HW + m0, 0, 3);
#if HAVE_TDM
    // async DMA: input tile (16 pixels x 32 channels) and weight tile
    // (32 k x 64 n) straight into LDS, then wait on TENSORcnt.
    tdm_load_2d_f32((unsigned)(uintptr_t)&Af[0], xi + (size_t)k0 * HW + m0,
                    16u, 32u, (unsigned)HW);
    tdm_load_2d_f32((unsigned)(uintptr_t)&Bf[0], w + (size_t)n0 * inC + k0,
                    32u, 64u, (unsigned)inC);
    __builtin_amdgcn_s_wait_tensorcnt(0);
    asm volatile("" ::: "memory");
#else
#pragma unroll
    for (int e = 0; e < 16; ++e) {
      int idx = e * 32 + l;
      int kk = idx >> 4, m = idx & 15;
      Af[kk * 16 + m] = xi[(size_t)(k0 + kk) * HW + m0 + m];
    }
#pragma unroll
    for (int e = 0; e < 64; ++e) {
      int idx = e * 32 + l;
      int n = idx >> 5, kk = idx & 31;
      Bf[n * 32 + kk] = w[(size_t)(n0 + n) * inC + k0 + kk];
    }
    __syncthreads();
#endif
    v16h a = frag_a_f32t(Af);
#pragma unroll
    for (int nt = 0; nt < 4; ++nt) {
      v16h b = frag_b_f32n(Bf, nt * 16);
      acc[nt] = wmma16(a, b, acc[nt]);
    }
#if HAVE_TDM
    // LDS reads must retire before next TDM overwrites the tiles.
    asm volatile("s_wait_dscnt 0" ::: "memory");
#else
    __syncthreads();
#endif
  }
  const int hi = l >> 4, lo = l & 15;
#pragma unroll
  for (int nt = 0; nt < 4; ++nt) {
    const int n = n0 + nt * 16 + lo;
    const float sa = s[n], ta = t[n];
#pragma unroll
    for (int i = 0; i < 8; ++i) {
      int m = m0 + i + 8 * hi;
      yi[(size_t)n * HW + m] = silu_f(acc[nt][i] * sa + ta);
    }
  }
}

// ---------------------------------------------------------------------------
// 3x3 SAME conv (128->128) as 9-tap GEMM accumulation, 16(M) x 64(N) block
// per wave; epilogue silu(bn1) then relu(bn_e). Border zeroing element-wise,
// so tiles are staged manually.
// ---------------------------------------------------------------------------
__global__ void fm_conv3x3(const float* __restrict__ x,
                           const float* __restrict__ w,
                           const float* __restrict__ s1,
                           const float* __restrict__ t1,
                           const float* __restrict__ s2,
                           const float* __restrict__ t2,
                           float* __restrict__ y) {
  __shared__ _Float16 As[16 * 32];   // [m][k]
  __shared__ _Float16 Bs[32 * 64];   // [k][n]
  const int l   = threadIdx.x;
  const int m0  = blockIdx.x * 16;
  const int n0  = blockIdx.y * 64;
  const int img = blockIdx.z;
  const float* xi = x + (size_t)img * 128 * HW;
  float* yi = y + (size_t)img * 128 * HW;

  v8f acc[4];
#pragma unroll
  for (int nt = 0; nt < 4; ++nt)
#pragma unroll
    for (int i = 0; i < 8; ++i) acc[nt][i] = 0.f;

  for (int tap = 0; tap < 9; ++tap) {
    const int dy = tap / 3 - 1, dx = tap % 3 - 1;
    for (int k0 = 0; k0 < 128; k0 += 32) {
#pragma unroll
      for (int e = 0; e < 16; ++e) {
        int idx = e * 32 + l;
        int m = idx >> 5, kk = idx & 31;
        int p = m0 + m;
        int py = (p >> 6) + dy, px = (p & 63) + dx;
        float v = 0.f;
        if ((unsigned)py < 64u && (unsigned)px < 64u)
          v = xi[(size_t)(k0 + kk) * HW + py * 64 + px];
        As[m * 32 + kk] = (_Float16)v;
      }
#pragma unroll
      for (int e = 0; e < 64; ++e) {
        int idx = e * 32 + l;
        int kb = idx >> 6, n = idx & 63;
        Bs[kb * 64 + n] = (_Float16)w[((size_t)(n0 + n) * 128 + k0 + kb) * 9 + tap];
      }
      __syncthreads();
      v16h a = frag_a_h(As);
#pragma unroll
      for (int nt = 0; nt < 4; ++nt) {
        v16h b = frag_b_h64(Bs, nt * 16);
        acc[nt] = wmma16(a, b, acc[nt]);
      }
      __syncthreads();
    }
  }
  const int hi = l >> 4, lo = l & 15;
#pragma unroll
  for (int nt = 0; nt < 4; ++nt) {
    const int n = n0 + nt * 16 + lo;
    const float sa = s1[n], ta = t1[n], sb = s2[n], tb = t2[n];
#pragma unroll
    for (int i = 0; i < 8; ++i) {
      int m = m0 + i + 8 * hi;
      float v = silu_f(acc[nt][i] * sa + ta);
      v = v * sb + tb;
      yi[(size_t)n * HW + m] = v > 0.f ? v : 0.f;
    }
  }
}

// ---------------------------------------------------------------------------
// Per-pixel L2 response over 64 channels + f16 normalized descriptors.
// ---------------------------------------------------------------------------
__global__ void fm_resp_norm(const float* __restrict__ desc,
                             float* __restrict__ resp,
                             _Float16* __restrict__ dn) {
  const int img = blockIdx.y;
  const int p   = blockIdx.x * 256 + threadIdx.x;
  const float* di = desc + (size_t)img * 64 * HW;
  float ss = 0.f;
#pragma unroll
  for (int c = 0; c < 64; ++c) {
    float v = di[(size_t)c * HW + p];
    ss += v * v;
  }
  float r = sqrtf(ss);
  resp[(size_t)img * HW + p] = r;
  float inv = 1.f / fmaxf(r, 1e-12f);
  _Float16* dni = dn + (size_t)img * 64 * HW;
#pragma unroll
  for (int c = 0; c < 64; ++c)
    dni[(size_t)c * HW + p] = (_Float16)(di[(size_t)c * HW + p] * inv);
}

// ---------------------------------------------------------------------------
// Top-512 of 4096 via in-LDS bitonic sort (descending, ties -> lower index),
// matching jax.lax.top_k. key = (valbits<<32)|(4095-idx), resp >= 0.
// ---------------------------------------------------------------------------
__global__ void fm_topk(const float* __restrict__ resp, int* __restrict__ idxout) {
  __shared__ unsigned long long keys[4096];
  const int img = blockIdx.x;
  const float* r = resp + (size_t)img * HW;
  for (int i = threadIdx.x; i < 4096; i += 256) {
    unsigned int vb = __float_as_uint(r[i]);
    keys[i] = ((unsigned long long)vb << 32) | (unsigned)(4095 - i);
  }
  __syncthreads();
  for (int k2 = 2; k2 <= 4096; k2 <<= 1) {
    for (int j = k2 >> 1; j > 0; j >>= 1) {
      for (int i = threadIdx.x; i < 4096; i += 256) {
        int l2 = i ^ j;
        if (l2 > i) {
          unsigned long long a = keys[i], b = keys[l2];
          bool asc = ((i & k2) != 0);  // overall descending
          bool sw  = asc ? (a > b) : (a < b);
          if (sw) { keys[i] = b; keys[l2] = a; }
        }
      }
      __syncthreads();
    }
  }
  for (int k = threadIdx.x; k < 512; k += 256)
    idxout[img * 512 + k] = 4095 - (int)(keys[k] & 0xFFFull);
}

// ---------------------------------------------------------------------------
// Gather top-k rgb descriptor rows: g1[b][k][c] = dn_rgb[b][c][idx1[k]]
// ---------------------------------------------------------------------------
__global__ void fm_gather_g1(const _Float16* __restrict__ dn,
                             const int* __restrict__ idxbuf,
                             _Float16* __restrict__ g1) {
  const int b = blockIdx.y;
  const int t = blockIdx.x * 256 + threadIdx.x;  // 512*64
  const int k = t >> 6, c = t & 63;
  const int p = idxbuf[b * 512 + k];
  g1[((size_t)b * 512 + k) * 64 + c] = dn[((size_t)b * 64 + c) * HW + p];
}

// ---------------------------------------------------------------------------
// Correlation rows for top-k keypoints: corr[b][k][m] = sum_c g1[b][k][c] *
// d2n[b][c][m].  WMMA GEMM, 16(keypoints) x 64(pixels) block per wave, K=64.
// ---------------------------------------------------------------------------
__global__ void fm_corr(const _Float16* __restrict__ g1,
                        const _Float16* __restrict__ dn,
                        float* __restrict__ corr) {
  __shared__ _Float16 As[16 * 32];   // [m][k]
  __shared__ _Float16 Bs[32 * 64];   // [k][n]
  const int l    = threadIdx.x;
  const int m0   = blockIdx.x * 16;  // keypoints
  const int col0 = blockIdx.y * 64;  // ir pixels
  const int b    = blockIdx.z;
  const _Float16* A = g1 + (size_t)b * 512 * 64;       // [512][64]
  const _Float16* B = dn + (size_t)(2 + b) * 64 * HW;  // ir image b: [64][4096]

  v8f acc[4];
#pragma unroll
  for (int nt = 0; nt < 4; ++nt)
#pragma unroll
    for (int i = 0; i < 8; ++i) acc[nt][i] = 0.f;

  for (int k0 = 0; k0 < 64; k0 += 32) {
#pragma unroll
    for (int e = 0; e < 16; ++e) {
      int idx = e * 32 + l;
      int m = idx >> 5, kk = idx & 31;
      As[m * 32 + kk] = A[(size_t)(m0 + m) * 64 + k0 + kk];
    }
#pragma unroll
    for (int e = 0; e < 64; ++e) {
      int idx = e * 32 + l;
      int kb = idx >> 6, n = idx & 63;
      Bs[kb * 64 + n] = B[(size_t)(k0 + kb) * HW + col0 + n];
    }
    __syncthreads();
    v16h a = frag_a_h(As);
#pragma unroll
    for (int nt = 0; nt < 4; ++nt) {
      v16h bfr = frag_b_h64(Bs, nt * 16);
      acc[nt] = wmma16(a, bfr, acc[nt]);
    }
    __syncthreads();
  }
  const int hi = l >> 4, lo = l & 15;
#pragma unroll
  for (int nt = 0; nt < 4; ++nt)
#pragma unroll
    for (int i = 0; i < 8; ++i) {
      int m = m0 + i + 8 * hi;
      corr[((size_t)b * 512 + m) * HW + col0 + nt * 16 + lo] = acc[nt][i];
    }
}

// ---------------------------------------------------------------------------
// Per-row softmax over 4096 ir pixels, gathered at column idx2[k].
// ---------------------------------------------------------------------------
__global__ void fm_softmax_gather(const float* __restrict__ corr,
                                  const int* __restrict__ idxbuf,
                                  float* __restrict__ wts) {
  __shared__ float red[256];
  const int bk = blockIdx.x;  // b*512+k
  const int b = bk >> 9, k = bk & 511;
  const float* row = corr + (size_t)bk * HW;
  float mx = -1e30f;
  for (int i = threadIdx.x; i < 4096; i += 256) mx = fmaxf(mx, row[i]);
  red[threadIdx.x] = mx;
  __syncthreads();
  for (int st = 128; st > 0; st >>= 1) {
    if (threadIdx.x < st)
      red[threadIdx.x] = fmaxf(red[threadIdx.x], red[threadIdx.x + st]);
    __syncthreads();
  }
  mx = red[0];
  __syncthreads();
  float sm = 0.f;
  for (int i = threadIdx.x; i < 4096; i += 256) sm += __expf(row[i] - mx);
  red[threadIdx.x] = sm;
  __syncthreads();
  for (int st = 128; st > 0; st >>= 1) {
    if (threadIdx.x < st) red[threadIdx.x] += red[threadIdx.x + st];
    __syncthreads();
  }
  if (threadIdx.x == 0) {
    int col = idxbuf[(2 + b) * 512 + k];
    wts[bk] = __expf(row[col] - mx) / red[0];
  }
}

// ---------------------------------------------------------------------------
// Weighted DLT: M = A^T A (9x9) via LDS float atomics, then one-thread
// cyclic Jacobi.  Reference H = row 8 of V, eigenvectors ordered by
// descending eigenvalue.
// ---------------------------------------------------------------------------
__global__ void fm_dlt(const int* __restrict__ idxbuf,
                       const float* __restrict__ wts, float* __restrict__ Hout) {
  __shared__ float M[81];
  const int b = blockIdx.x;
  if (threadIdx.x < 81) M[threadIdx.x] = 0.f;
  __syncthreads();

  for (int k = threadIdx.x; k < 512; k += 256) {
    const int i1 = idxbuf[b * 512 + k];
    const int i2 = idxbuf[(2 + b) * 512 + k];
    const float w  = wts[b * 512 + k];
    const float w2 = w * w;
    const float x1 = (float)(i1 & 63) / 63.f * 2.f - 1.f;
    const float y1 = (float)(i1 >> 6) / 63.f * 2.f - 1.f;
    const float x2 = (float)(i2 & 63) / 63.f * 2.f - 1.f;
    const float y2 = (float)(i2 >> 6) / 63.f * 2.f - 1.f;
    const float r1[9] = {x1, y1, 1.f, 0.f, 0.f, 0.f, -x2 * x1, -x2 * y1, -x2};
    const float r2[9] = {0.f, 0.f, 0.f, x1, y1, 1.f, -y2 * x1, -y2 * y1, -y2};
#pragma unroll
    for (int i = 0; i < 9; ++i)
#pragma unroll
      for (int j = 0; j < 9; ++j)
        atomicAdd(&M[i * 9 + j], w2 * (r1[i] * r1[j] + r2[i] * r2[j]));
  }
  __syncthreads();

  if (threadIdx.x == 0) {
    float A[81], V[81];
    for (int i = 0; i < 81; ++i) { A[i] = M[i]; V[i] = 0.f; }
    for (int i = 0; i < 9; ++i) V[i * 9 + i] = 1.f;
    for (int sweep = 0; sweep < 24; ++sweep) {
      for (int p = 0; p < 8; ++p) {
        for (int q = p + 1; q < 9; ++q) {
          float apq = A[p * 9 + q];
          if (fabsf(apq) < 1e-14f) continue;
          float app = A[p * 9 + p], aqq = A[q * 9 + q];
          float th = 0.5f * (aqq - app) / apq;
          float tt = (th >= 0.f ? 1.f : -1.f) /
                     (fabsf(th) + sqrtf(th * th + 1.f));
          float c = 1.f / sqrtf(tt * tt + 1.f);
          float s = tt * c;
          for (int i = 0; i < 9; ++i) {
            float aip = A[i * 9 + p], aiq = A[i * 9 + q];
            A[i * 9 + p] = c * aip - s * aiq;
            A[i * 9 + q] = s * aip + c * aiq;
          }
          for (int i = 0; i < 9; ++i) {
            float api = A[p * 9 + i], aqi = A[q * 9 + i];
            A[p * 9 + i] = c * api - s * aqi;
            A[q * 9 + i] = s * api + c * aqi;
          }
          for (int i = 0; i < 9; ++i) {
            float vip = V[i * 9 + p], viq = V[i * 9 + q];
            V[i * 9 + p] = c * vip - s * viq;
            V[i * 9 + q] = s * vip + c * viq;
          }
        }
      }
    }
    int ord[9];
    for (int i = 0; i < 9; ++i) ord[i] = i;
    for (int i = 0; i < 9; ++i)
      for (int j = i + 1; j < 9; ++j)
        if (A[ord[j] * 9 + ord[j]] > A[ord[i] * 9 + ord[i]]) {
          int tmp = ord[i]; ord[i] = ord[j]; ord[j] = tmp;
        }
    for (int j = 0; j < 9; ++j) Hout[b * 9 + j] = V[8 * 9 + ord[j]];
  }
}

// ---------------------------------------------------------------------------
extern "C" void kernel_launch(void* const* d_in, const int* in_sizes, int n_in,
                              void* d_out, int out_size, void* d_ws,
                              size_t ws_size, hipStream_t stream) {
  (void)in_sizes; (void)n_in; (void)out_size; (void)ws_size;
  const float* rgb  = (const float*)d_in[0];
  const float* ir   = (const float*)d_in[1];
  const float* w_dr = (const float*)d_in[2];
  const float* g_dr = (const float*)d_in[3];
  const float* b_dr = (const float*)d_in[4];
  const float* m_dr = (const float*)d_in[5];
  const float* v_dr = (const float*)d_in[6];
  const float* w_d1 = (const float*)d_in[7];
  const float* g_d1 = (const float*)d_in[8];
  const float* b_d1 = (const float*)d_in[9];
  const float* m_d1 = (const float*)d_in[10];
  const float* v_d1 = (const float*)d_in[11];
  const float* g_e  = (const float*)d_in[12];
  const float* b_e  = (const float*)d_in[13];
  const float* m_e  = (const float*)d_in[14];
  const float* v_e  = (const float*)d_in[15];
  const float* w_d2 = (const float*)d_in[16];
  const float* g_d2 = (const float*)d_in[17];
  const float* b_d2 = (const float*)d_in[18];
  const float* m_d2 = (const float*)d_in[19];
  const float* v_d2 = (const float*)d_in[20];

  float* out = (float*)d_out;   // [H(18)] [rgb_desc 2*64*4096] [ir_desc 2*64*4096]
  float* ws  = (float*)d_ws;

  float* s_dr = ws + 0;   float* t_dr = ws + 128;
  float* s_d1 = ws + 256; float* t_d1 = ws + 384;
  float* s_e  = ws + 512; float* t_e  = ws + 640;
  float* s_d2 = ws + 768; float* t_d2 = ws + 832;
  float* h1   = ws + 1024;                       // 4*128*4096 f32
  float* h2   = h1 + (size_t)4 * 128 * 4096;     // 4*128*4096 f32
  float* resp = h2 + (size_t)4 * 128 * 4096;     // 4*4096 f32
  float* corr = resp + (size_t)4 * 4096;         // 2*512*4096 f32
  float* wts  = corr + (size_t)2 * 512 * 4096;   // 1024 f32
  _Float16* dn = (_Float16*)(wts + 1024);        // 4*64*4096 f16
  _Float16* g1 = dn + (size_t)4 * 64 * 4096;     // 2*512*64 f16
  int* idxbuf  = (int*)(g1 + (size_t)2 * 512 * 64);  // 4*512 int

  fm_prep_bn<<<1, 128, 0, stream>>>(g_dr, b_dr, m_dr, v_dr, s_dr, t_dr, 128);
  fm_prep_bn<<<1, 128, 0, stream>>>(g_d1, b_d1, m_d1, v_d1, s_d1, t_d1, 128);
  fm_prep_bn<<<1, 128, 0, stream>>>(g_e,  b_e,  m_e,  v_e,  s_e,  t_e,  128);
  fm_prep_bn<<<1, 128, 0, stream>>>(g_d2, b_d2, m_d2, v_d2, s_d2, t_d2, 64);

  float* desc = out + 18;  // imgs: 0,1 = rgb_desc b0,b1 ; 2,3 = ir_desc b0,b1

  // dim_reduce: 256 -> 128, 1x1 (TDM-staged GEMM)
  fm_conv1x1<<<dim3(256, 2, 4), 32, 0, stream>>>(rgb, ir, w_dr, s_dr, t_dr, h1,
                                                 256, 128);
  // descriptor conv1: 128 -> 128, 3x3 SAME, silu(bn) then relu(bn_e)
  fm_conv3x3<<<dim3(256, 2, 4), 32, 0, stream>>>(h1, w_d1, s_d1, t_d1, s_e, t_e,
                                                 h2);
  // descriptor conv2: 128 -> 64, 1x1, writes descriptors straight into d_out
  fm_conv1x1<<<dim3(256, 1, 4), 32, 0, stream>>>(
      h2, h2 + (size_t)2 * 128 * 4096, w_d2, s_d2, t_d2, desc, 128, 64);

  fm_resp_norm<<<dim3(16, 4), 256, 0, stream>>>(desc, resp, dn);
  fm_topk<<<4, 256, 0, stream>>>(resp, idxbuf);
  fm_gather_g1<<<dim3(128, 2), 256, 0, stream>>>(dn, idxbuf, g1);
  fm_corr<<<dim3(32, 64, 2), 32, 0, stream>>>(g1, dn, corr);
  fm_softmax_gather<<<1024, 256, 0, stream>>>(corr, idxbuf, wts);
  fm_dlt<<<2, 256, 0, stream>>>(idxbuf, wts, out);
}
